// SingleMP_Tension_3427383902968
// MI455X (gfx1250) — matrix operand order repaired and checked by reading, not compiled, not run on hardware
//
#include <hip/hip_runtime.h>

#define NN 50000
#define EE 500000

typedef __attribute__((ext_vector_type(16))) _Float16 v16h;
typedef __attribute__((ext_vector_type(8)))  _Float16 h8;
typedef __attribute__((ext_vector_type(8)))  float    v8f;

// ---------------------------------------------------------------------------
// A-fragment loader: 16x32 f16 A tile from LDS (row-major, strideH halves).
// ISA lane layout (CDNA5 ISA, 16-bit A 16x32):
//   lane<16 : row=lane, elems 0..7 = K kb+0..7,  elems 8..15 = K kb+16..23
//   lane>=16: row=lane-16, elems 0..7 = K kb+8..15, elems 8..15 = K kb+24..31
// ---------------------------------------------------------------------------
__device__ __forceinline__ v16h load_a_frag(const _Float16* base, int strideH,
                                            int kb, int lane) {
  const int row = lane & 15;
  const int hi  = (lane >> 4) & 1;
  const _Float16* p = base + row * strideH + kb + hi * 8;
  union { v16h v; h8 h[2]; } u;
  u.h[0] = *(const h8*)(p);
  u.h[1] = *(const h8*)(p + 16);
  return u.v;
}

// B-fragment: pre-swizzled in workspace; lane reads 16 contiguous halves (32B).
__device__ __forceinline__ v16h load_b_frag(const _Float16* packed, int frag,
                                            int lane) {
  return *(const v16h*)(packed + ((long)frag * 32 + lane) * 16);
}

__device__ __forceinline__ v8f wmma16(v16h a, v16h b, v8f c) {
  return __builtin_amdgcn_wmma_f32_16x16x32_f16(false, a, false, b,
                                                (short)0, c, false, false);
}

// ---------------------------------------------------------------------------
// Pack weight W[K x Nc] (f32, row-major) into WMMA-B f16 fragments.
// Fragment f = kstep*ntiles + ntile; within fragment, lane l holds
// K = kstep*32 + (j<8 ? (l>=16)*8 + j : 16 + (l>=16)*8 + (j-8)) (j=0..15), N = ntile*16 + (l&15).
// Rows k>=K are zero padding (Kpad is a multiple of 32).
// ---------------------------------------------------------------------------
__global__ void pack_w_kernel(const float* __restrict__ W, int K, int Kpad,
                              int Nc, _Float16* __restrict__ out) {
  int t = blockIdx.x * blockDim.x + threadIdx.x;
  int total = (Kpad / 32) * (Nc / 16) * 512;
  if (t >= total) return;
  int f = t >> 9;
  int l = (t >> 4) & 31;
  int j = t & 15;
  int ntiles = Nc / 16;
  int ks = f / ntiles, nt = f % ntiles;
  int k = ks * 32 + ((l >> 4) & 1) * 8 + (j & 7) + (j >> 3) * 16;
  int n = nt * 16 + (l & 15);
  float v = (k < K) ? W[(long)k * Nc + n] : 0.0f;
  out[t] = (_Float16)v;
}

__global__ void zero_kernel(float* __restrict__ p, long n) {
  long i = (long)blockIdx.x * blockDim.x + threadIdx.x;
  if (i < n) p[i] = 0.0f;
}

// ---------------------------------------------------------------------------
// Edge kernel: one wave = 8 edge-pairs = 16 directed-edge rows.
// Rows 0..7  = forward edges  (diff = x[t]-x[s],  +ea) -> scatter to tgt
// Rows 8..15 = reversed edges (negated input)          -> scatter to src
// layer1: [16x96]@[96x128] (K padded 72->96); layer2: [16x128]@[128x64]
// tension: rows r + r+8 summed -> [16x64]@[64x32] -> dot with Wt2.
// ---------------------------------------------------------------------------
__global__ __launch_bounds__(128)
void edge_kernel(const float* __restrict__ x, const int* __restrict__ ei,
                 const float* __restrict__ ea,
                 const _Float16* __restrict__ pWm1, const float* __restrict__ bm1,
                 const _Float16* __restrict__ pWm2, const float* __restrict__ bm2,
                 const _Float16* __restrict__ pWt1, const float* __restrict__ bt1,
                 const float* __restrict__ wt2, const float* __restrict__ bt2,
                 float* __restrict__ sums, float* __restrict__ cnt,
                 float* __restrict__ e_out) {
  __shared__ __align__(32) _Float16 ldsA[4][16 * 104];  // input tile, K pad 96
  __shared__ __align__(32) _Float16 ldsH[4][16 * 136];  // hidden 128, pad 136
  __shared__ __align__(32) _Float16 ldsM[4][16 * 72];   // messages 64, pad 72
  __shared__ __align__(32) float    ldsT[4][16 * 36];   // tension hidden 32

  const int lane = threadIdx.x & 31;
  const int wv   = threadIdx.x >> 5;
  const int ep0  = ((int)blockIdx.x * 4 + wv) * 8;  // first edge-pair of wave
  if (ep0 >= EE) return;

  _Float16* A = ldsA[wv];
  _Float16* H = ldsH[wv];
  _Float16* M = ldsM[wv];
  float*    T = ldsT[wv];

  const int hi  = (lane >> 4) & 1;
  const int nlo = lane & 15;

  int sv[8], tv[8];
#pragma unroll
  for (int i = 0; i < 8; i++) {
    sv[i] = ei[ep0 + i];        // edge_index[0][e]  (src)
    tv[i] = ei[EE + ep0 + i];   // edge_index[1][e]  (tgt)
  }

  // ---- build input tile: row r = [x[t]-x[s] | ea | 0pad], row r+8 = -row r
#pragma unroll
  for (int r = 0; r < 8; r++) {
    const int s = sv[r], t = tv[r];
    for (int c = lane; c < 104; c += 32) {
      float v;
      if (c < 64)       v = x[(long)t * 64 + c] - x[(long)s * 64 + c];
      else if (c < 72)  v = ea[(long)(ep0 + r) * 8 + (c - 64)];
      else              v = 0.0f;
      A[r * 104 + c]       = (_Float16)v;
      A[(r + 8) * 104 + c] = (_Float16)(-v);
    }
  }
  __builtin_amdgcn_wave_barrier();

  // ---- layer 1: h = relu(in @ Wm1 + bm1), 3 K-steps x 8 N-tiles
  v8f acc1[8] = {};
#pragma unroll
  for (int ks = 0; ks < 3; ks++) {
    v16h a = load_a_frag(A, 104, ks * 32, lane);
#pragma unroll
    for (int nt = 0; nt < 8; nt++)
      acc1[nt] = wmma16(a, load_b_frag(pWm1, ks * 8 + nt, lane), acc1[nt]);
  }
#pragma unroll
  for (int nt = 0; nt < 8; nt++) {
    float bias = bm1[nt * 16 + nlo];
#pragma unroll
    for (int v = 0; v < 8; v++) {
      float val = acc1[nt][v] + bias;
      val = val > 0.0f ? val : 0.0f;
      H[(v + 8 * hi) * 136 + nt * 16 + nlo] = (_Float16)val;
    }
  }
  __builtin_amdgcn_wave_barrier();

  // ---- layer 2: m = relu(h @ Wm2 + bm2), 4 K-steps x 4 N-tiles
  v8f acc2[4] = {};
#pragma unroll
  for (int ks = 0; ks < 4; ks++) {
    v16h a = load_a_frag(H, 136, ks * 32, lane);
#pragma unroll
    for (int nt = 0; nt < 4; nt++)
      acc2[nt] = wmma16(a, load_b_frag(pWm2, ks * 4 + nt, lane), acc2[nt]);
  }
  // stash m for tension + atomic scatter-add into sums[node][n]
#pragma unroll
  for (int nt = 0; nt < 4; nt++) {
    float bias = bm2[nt * 16 + nlo];
#pragma unroll
    for (int v = 0; v < 8; v++) {
      float val = acc2[nt][v] + bias;
      val = val > 0.0f ? val : 0.0f;
      M[(v + 8 * hi) * 72 + nt * 16 + nlo] = (_Float16)val;
      const int node = hi ? sv[v] : tv[v];  // bwd rows -> src, fwd rows -> tgt
      atomicAdd(&sums[(long)node * 64 + nt * 16 + nlo], val);
    }
  }
  // degree counts (one +1 per directed edge at its target)
  if (lane < 16) {
    const int e = ep0 + (lane & 7);
    const int node = (lane < 8) ? ei[EE + e] : ei[e];
    atomicAdd(&cnt[node], 1.0f);
  }
  __builtin_amdgcn_wave_barrier();

  // ---- tension: msym rows 0..7 = m[r] + m[r+8]; rows 8..15 unused garbage
  for (int c = lane; c < 64; c += 32) {
#pragma unroll
    for (int r = 0; r < 8; r++)
      A[r * 104 + c] =
          (_Float16)((float)M[r * 72 + c] + (float)M[(r + 8) * 72 + c]);
  }
  __builtin_amdgcn_wave_barrier();

  v8f acc3[2] = {};
#pragma unroll
  for (int ks = 0; ks < 2; ks++) {
    v16h a = load_a_frag(A, 104, ks * 32, lane);
#pragma unroll
    for (int nt = 0; nt < 2; nt++)
      acc3[nt] = wmma16(a, load_b_frag(pWt1, ks * 2 + nt, lane), acc3[nt]);
  }
#pragma unroll
  for (int nt = 0; nt < 2; nt++) {
    float bias = bt1[nt * 16 + nlo];
#pragma unroll
    for (int v = 0; v < 8; v++) {
      float val = acc3[nt][v] + bias;
      T[(v + 8 * hi) * 36 + nt * 16 + nlo] = val > 0.0f ? val : 0.0f;
    }
  }
  __builtin_amdgcn_wave_barrier();

  if (lane < 8) {
    float s2 = bt2[0];
#pragma unroll
    for (int k = 0; k < 32; k++) s2 += T[lane * 36 + k] * wt2[k];
    e_out[ep0 + lane] = s2;
  }
}

// ---------------------------------------------------------------------------
// Node kernel: one wave = 16 nodes. A = [x | sums/max(cnt,1)] (K=128),
// hu = relu(A @ Wu1 + bu1); x_out = hu @ Wu2 + bu2 (no relu).
// ---------------------------------------------------------------------------
__global__ __launch_bounds__(128)
void node_kernel(const float* __restrict__ x, const float* __restrict__ sums,
                 const float* __restrict__ cnt,
                 const _Float16* __restrict__ pWu1, const float* __restrict__ bu1,
                 const _Float16* __restrict__ pWu2, const float* __restrict__ bu2,
                 float* __restrict__ x_out) {
  __shared__ __align__(32) _Float16 ldsA[4][16 * 136];
  __shared__ __align__(32) _Float16 ldsH[4][16 * 72];

  const int lane = threadIdx.x & 31;
  const int wv   = threadIdx.x >> 5;
  const long nb  = ((long)blockIdx.x * 4 + wv) * 16;
  if (nb >= NN) return;

  _Float16* A = ldsA[wv];
  _Float16* H = ldsH[wv];
  const int hi  = (lane >> 4) & 1;
  const int nlo = lane & 15;

  for (int r = 0; r < 16; r++) {
    const long node = nb + r;
    const float rinv = 1.0f / fmaxf(cnt[node], 1.0f);
    for (int c = lane; c < 136; c += 32) {
      float v;
      if (c < 64)        v = x[node * 64 + c];
      else if (c < 128)  v = sums[node * 64 + (c - 64)] * rinv;
      else               v = 0.0f;
      A[r * 136 + c] = (_Float16)v;
    }
  }
  __builtin_amdgcn_wave_barrier();

  v8f acc[4] = {};
#pragma unroll
  for (int ks = 0; ks < 4; ks++) {
    v16h a = load_a_frag(A, 136, ks * 32, lane);
#pragma unroll
    for (int nt = 0; nt < 4; nt++)
      acc[nt] = wmma16(a, load_b_frag(pWu1, ks * 4 + nt, lane), acc[nt]);
  }
#pragma unroll
  for (int nt = 0; nt < 4; nt++) {
    float bias = bu1[nt * 16 + nlo];
#pragma unroll
    for (int v = 0; v < 8; v++) {
      float val = acc[nt][v] + bias;
      val = val > 0.0f ? val : 0.0f;
      H[(v + 8 * hi) * 72 + nt * 16 + nlo] = (_Float16)val;
    }
  }
  __builtin_amdgcn_wave_barrier();

  v8f acc2[2] = {};
#pragma unroll
  for (int ks = 0; ks < 2; ks++) {
    v16h a = load_a_frag(H, 72, ks * 32, lane);
#pragma unroll
    for (int nt = 0; nt < 2; nt++)
      acc2[nt] = wmma16(a, load_b_frag(pWu2, ks * 2 + nt, lane), acc2[nt]);
  }
#pragma unroll
  for (int nt = 0; nt < 2; nt++) {
    float bias = bu2[nt * 16 + nlo];
#pragma unroll
    for (int v = 0; v < 8; v++)
      x_out[(nb + v + 8 * hi) * 32 + nt * 16 + nlo] = acc2[nt][v] + bias;
  }
}

// ---------------------------------------------------------------------------
extern "C" void kernel_launch(void* const* d_in, const int* in_sizes, int n_in,
                              void* d_out, int out_size, void* d_ws,
                              size_t ws_size, hipStream_t stream) {
  const float* x    = (const float*)d_in[0];
  const int*   ei   = (const int*)d_in[1];
  const float* ea   = (const float*)d_in[2];
  const float* Wm1  = (const float*)d_in[3];
  const float* bm1  = (const float*)d_in[4];
  const float* Wm2  = (const float*)d_in[5];
  const float* bm2  = (const float*)d_in[6];
  const float* Wu1  = (const float*)d_in[7];
  const float* bu1  = (const float*)d_in[8];
  const float* Wu2  = (const float*)d_in[9];
  const float* bu2  = (const float*)d_in[10];
  const float* Wt1  = (const float*)d_in[11];
  const float* bt1  = (const float*)d_in[12];
  const float* Wt2  = (const float*)d_in[13];
  const float* bt2  = (const float*)d_in[14];

  char* ws = (char*)d_ws;
  // workspace layout (all offsets 32B-aligned)
  float*    sums  = (float*)ws;                                 // N*64 f32
  float*    cnt   = (float*)(ws + (size_t)NN * 64 * 4);         // N f32
  size_t    off   = (size_t)NN * 64 * 4 + (size_t)NN * 4;       // 13,000,000
  _Float16* pWm1  = (_Float16*)(ws + off);  off += 3 * 8 * 512 * 2;  // 24576
  _Float16* pWm2  = (_Float16*)(ws + off);  off += 4 * 4 * 512 * 2;  // 16384
  _Float16* pWu1  = (_Float16*)(ws + off);  off += 4 * 4 * 512 * 2;  // 16384
  _Float16* pWu2  = (_Float16*)(ws + off);  off += 2 * 2 * 512 * 2;  // 4096
  _Float16* pWt1  = (_Float16*)(ws + off);  off += 2 * 2 * 512 * 2;  // 4096

  float* x_out = (float*)d_out;                 // [N, 32]
  float* e_out = x_out + (size_t)NN * 32;       // [E]

  const long nz = (long)NN * 64 + NN;  // sums + cnt contiguous
  zero_kernel<<<(int)((nz + 255) / 256), 256, 0, stream>>>(sums, nz);

  pack_w_kernel<<<48, 256, 0, stream>>>(Wm1, 72, 96, 128, pWm1);
  pack_w_kernel<<<32, 256, 0, stream>>>(Wm2, 128, 128, 64, pWm2);
  pack_w_kernel<<<32, 256, 0, stream>>>(Wu1, 128, 128, 64, pWu1);
  pack_w_kernel<<<8, 256, 0, stream>>>(Wu2, 64, 64, 32, pWu2);
  pack_w_kernel<<<8, 256, 0, stream>>>(Wt1, 64, 64, 32, pWt1);

  // EE / 8 edge-pairs per wave / 4 waves per block = 15625 blocks (exact)
  edge_kernel<<<EE / 8 / 4, 128, 0, stream>>>(
      x, ei, ea, pWm1, bm1, pWm2, bm2, pWt1, bt1, Wt2, bt2, sums, cnt, e_out);

  // NN / 16 nodes per wave = 3125 waves -> ceil(3125/4) = 782 blocks
  node_kernel<<<(3125 + 3) / 4, 128, 0, stream>>>(x, sums, cnt, pWu1, bu1,
                                                  pWu2, bu2, x_out);
}
